// TimesFormerLayer_53712861004305
// MI455X (gfx1250) — matrix operand order repaired and checked by reading, not compiled, not run on hardware
//
#include <hip/hip_runtime.h>
#include <hip/hip_bf16.h>

// ---------------- constants ----------------
#define Bb   8
#define Hh   12
#define Tt   8
#define Pp   196
#define Ss   1569          // 1 + T*P
#define Dd   768
#define Dh   64
#define Dff  3072
#define Mtot (Bb*Ss)       // 12552

typedef __attribute__((ext_vector_type(16))) __bf16 v16bf;
typedef __attribute__((ext_vector_type(8)))  float  v8f;

#define LDT 40             // LDS tile row stride (bf16): 80B, 16B aligned, all-64-banks mapping

// ---------------- WMMA fragment loads (CDNA5 16-bit layouts) ----------------
// A 16x32: lanes 0-15 -> row m, K 0..7 & 16..23 ; lanes 16-31 -> K 8..15 & 24..31
__device__ __forceinline__ v16bf frag_a(const __bf16* s, int row, int hi) {
  union { uint4 u[2]; v16bf v; } r;
  const __bf16* p = s + row * LDT;
  r.u[0] = *reinterpret_cast<const uint4*>(p + hi * 8);
  r.u[1] = *reinterpret_cast<const uint4*>(p + 16 + hi * 8);
  return r.v;
}
// B 32x16 (held as Bt rows = N, contiguous K): lanes 0-15 -> col n, K 0..7 & 16..23 ; lanes 16-31 -> K 8..15 & 24..31
__device__ __forceinline__ v16bf frag_b(const __bf16* s, int row, int hi) {
  union { uint4 u[2]; v16bf v; } r;
  const __bf16* p = s + row * LDT;
  r.u[0] = *reinterpret_cast<const uint4*>(p + hi * 8);
  r.u[1] = *reinterpret_cast<const uint4*>(p + 16 + hi * 8);
  return r.v;
}

__device__ __forceinline__ v8f wmma_bf16(v16bf a, v16bf b, v8f c) {
  return __builtin_amdgcn_wmma_f32_16x16x32_bf16(false, a, false, b, (short)0, c, false, false);
}

// ---------------- GEMM: C[MxN] = A(bf16,MxK) * Bt(bf16,NxK)^T + bias (+epilogue) ----------------
// epi: 0 = bias -> f32 C ; 1 = bias + residual R -> f32 C ; 2 = bias + exact GELU -> bf16 C
// 128x128 block tile, 8 waves x (32x64), BK=32, double-buffered LDS, reg-prefetch pipeline.
__global__ void __launch_bounds__(256) gemm_bf16_kernel(
    const __bf16* __restrict__ A, const __bf16* __restrict__ Bt,
    const float* __restrict__ bias, const float* __restrict__ R,
    void* __restrict__ Cv, int M, int N, int K, int epi)
{
  __shared__ __align__(16) __bf16 sA[2][128 * LDT];
  __shared__ __align__(16) __bf16 sB[2][128 * LDT];

  const int tid    = threadIdx.x;
  const int lane16 = tid & 15;
  const int hi     = (tid >> 4) & 1;
  const int wave   = tid >> 5;
  const int wr     = wave >> 1;       // 0..3 : row group of 32
  const int wc     = wave & 1;        // 0..1 : col group of 64
  const int m0     = blockIdx.x * 128;
  const int n0     = blockIdx.y * 128;

  // staging: each thread copies 2x 16B chunks of A and of B per K-slab.
  // A rows are CLAMPED to M-1 (out-of-range rows are never stored -> contents don't-care),
  // keeping the loads unconditional global_load_b128.
  const int r0 = tid >> 2;            // 0..63
  const int c8 = (tid & 3) << 3;      // 0,8,16,24
  const int ra0 = min(m0 + r0,      M - 1);
  const int ra1 = min(m0 + r0 + 64, M - 1);
  const __bf16* aptr0 = A  + (size_t)ra0 * K + c8;
  const __bf16* aptr1 = A  + (size_t)ra1 * K + c8;
  const __bf16* bptr0 = Bt + (size_t)(n0 + r0)      * K + c8;
  const __bf16* bptr1 = Bt + (size_t)(n0 + r0 + 64) * K + c8;
  const int offA0 = r0 * LDT + c8;
  const int offA1 = (r0 + 64) * LDT + c8;

  uint4 pa0 = *reinterpret_cast<const uint4*>(aptr0);
  uint4 pa1 = *reinterpret_cast<const uint4*>(aptr1);
  uint4 pb0 = *reinterpret_cast<const uint4*>(bptr0);
  uint4 pb1 = *reinterpret_cast<const uint4*>(bptr1);

  v8f acc[2][4];
  #pragma unroll
  for (int i = 0; i < 2; i++)
    #pragma unroll
    for (int j = 0; j < 4; j++)
      acc[i][j] = (v8f){0.f,0.f,0.f,0.f,0.f,0.f,0.f,0.f};

  const int nk = K >> 5;              // number of 32-wide K slabs

  // stage slab 0 into buffer 0
  *reinterpret_cast<uint4*>(sA[0] + offA0) = pa0;
  *reinterpret_cast<uint4*>(sA[0] + offA1) = pa1;
  *reinterpret_cast<uint4*>(sB[0] + offA0) = pb0;
  *reinterpret_cast<uint4*>(sB[0] + offA1) = pb1;

  // prefetch slab 1 into registers
  if (nk > 1) {
    aptr0 += 32; aptr1 += 32; bptr0 += 32; bptr1 += 32;
    pa0 = *reinterpret_cast<const uint4*>(aptr0);
    pa1 = *reinterpret_cast<const uint4*>(aptr1);
    pb0 = *reinterpret_cast<const uint4*>(bptr0);
    pb1 = *reinterpret_cast<const uint4*>(bptr1);
  }

  int cur = 0;
  for (int k = 0; k < nk; k++) {
    __syncthreads();                  // buf[cur] writes visible; buf[cur^1] reads done

    const __bf16* cA = sA[cur];
    const __bf16* cB = sB[cur];
    v16bf a0 = frag_a(cA, wr * 32 + lane16,      hi);
    v16bf a1 = frag_a(cA, wr * 32 + 16 + lane16, hi);
    v16bf bfr[4];
    #pragma unroll
    for (int j = 0; j < 4; j++)
      bfr[j] = frag_b(cB, wc * 64 + j * 16 + lane16, hi);

    if (k + 1 < nk) {
      // store prefetched slab k+1 into the other buffer (overlaps WMMAs below)
      int nxt = cur ^ 1;
      *reinterpret_cast<uint4*>(sA[nxt] + offA0) = pa0;
      *reinterpret_cast<uint4*>(sA[nxt] + offA1) = pa1;
      *reinterpret_cast<uint4*>(sB[nxt] + offA0) = pb0;
      *reinterpret_cast<uint4*>(sB[nxt] + offA1) = pb1;
      if (k + 2 < nk) {               // prefetch slab k+2
        aptr0 += 32; aptr1 += 32; bptr0 += 32; bptr1 += 32;
        pa0 = *reinterpret_cast<const uint4*>(aptr0);
        pa1 = *reinterpret_cast<const uint4*>(aptr1);
        pb0 = *reinterpret_cast<const uint4*>(bptr0);
        pb1 = *reinterpret_cast<const uint4*>(bptr1);
      }
    }

    #pragma unroll
    for (int j = 0; j < 4; j++) {
      acc[0][j] = wmma_bf16(a0, bfr[j], acc[0][j]);
      acc[1][j] = wmma_bf16(a1, bfr[j], acc[1][j]);
    }
    cur ^= 1;
  }

  // epilogue: C/D layout -> M = hi*8+e, N = lane16 within each 16x16
  float*  Cf = reinterpret_cast<float*>(Cv);
  __bf16* Cb = reinterpret_cast<__bf16*>(Cv);
  #pragma unroll
  for (int i = 0; i < 2; i++) {
    #pragma unroll
    for (int j = 0; j < 4; j++) {
      int gc = n0 + wc * 64 + j * 16 + lane16;
      float bb = bias[gc];
      #pragma unroll
      for (int e = 0; e < 8; e++) {
        int gr = m0 + wr * 32 + i * 16 + hi * 8 + e;
        if (gr < M) {
          float v = acc[i][j][e] + bb;
          if (epi == 1) {
            v += R[(size_t)gr * N + gc];
            Cf[(size_t)gr * N + gc] = v;
          } else if (epi == 2) {
            v = 0.5f * v * (1.0f + erff(v * 0.70710678118654752f));
            Cb[(size_t)gr * N + gc] = (__bf16)v;
          } else {
            Cf[(size_t)gr * N + gc] = v;
          }
        }
      }
    }
  }
}

// ---------------- weight transpose + f32->bf16 : Wt[n*K+k] = bf16(W[k*N+n]) ----------------
__global__ void wcvt_kernel(const float* __restrict__ W, __bf16* __restrict__ Wt, int K, int N) {
  __shared__ float tile[32][33];
  int k0 = blockIdx.x * 32, n0 = blockIdx.y * 32;
  int tx = threadIdx.x, ty = threadIdx.y;          // (32,8)
  for (int i = ty; i < 32; i += 8)
    tile[i][tx] = W[(size_t)(k0 + i) * N + n0 + tx];
  __syncthreads();
  for (int i = ty; i < 32; i += 8)
    Wt[(size_t)(n0 + i) * K + k0 + tx] = (__bf16)tile[tx][i];
}

// ---------------- LayerNorm (row of 768) -> bf16, biased variance, eps=1e-12 ----------------
__global__ void __launch_bounds__(256) ln_kernel(const float* __restrict__ x,
                                                 const float* __restrict__ g,
                                                 const float* __restrict__ bt,
                                                 __bf16* __restrict__ y)
{
  int row = blockIdx.x, tid = threadIdx.x;
  const float* xr = x + (size_t)row * Dd;
  float v[3];
  #pragma unroll
  for (int i = 0; i < 3; i++) v[i] = xr[tid + 256 * i];
  float s = v[0] + v[1] + v[2];
  float q = v[0]*v[0] + v[1]*v[1] + v[2]*v[2];
  __shared__ float rs[256], rq[256];
  rs[tid] = s; rq[tid] = q; __syncthreads();
  for (int off = 128; off > 0; off >>= 1) {
    if (tid < off) { rs[tid] += rs[tid + off]; rq[tid] += rq[tid + off]; }
    __syncthreads();
  }
  float mean = rs[0] * (1.f / Dd);
  float var  = rq[0] * (1.f / Dd) - mean * mean;
  float rstd = rsqrtf(var + 1e-12f);
  __bf16* yr = y + (size_t)row * Dd;
  #pragma unroll
  for (int i = 0; i < 3; i++) {
    int c = tid + 256 * i;
    yr[c] = (__bf16)((v[i] - mean) * rstd * g[c] + bt[c]);
  }
}

// ---------------- elementwise copy (float4) ----------------
__global__ void copy_kernel(const float* __restrict__ in, float* __restrict__ out, int n4) {
  int i = blockIdx.x * 256 + threadIdx.x;
  if (i < n4) reinterpret_cast<float4*>(out)[i] = reinterpret_cast<const float4*>(in)[i];
}

// ---------------- cls-token attention: one block per (b,h), keys over full S ----------------
__global__ void __launch_bounds__(256) attn_cls_kernel(const float* __restrict__ Q,
                                                       const float* __restrict__ K,
                                                       const float* __restrict__ V,
                                                       __bf16* __restrict__ CTX)
{
  int b = blockIdx.x / Hh, h = blockIdx.x % Hh;
  int tid = threadIdx.x;
  __shared__ float sq[Dh];
  __shared__ float sc[Ss];
  __shared__ float red[256];
  const size_t base = ((size_t)b * Ss) * Dd + (size_t)h * Dh;

  if (tid < Dh) sq[tid] = Q[base + tid];
  __syncthreads();

  float mx = -1e30f;
  for (int k = tid; k < Ss; k += 256) {
    const float* kp = K + base + (size_t)k * Dd;
    float s = 0.f;
    #pragma unroll
    for (int d = 0; d < Dh; d++) s += sq[d] * kp[d];
    s *= 0.125f;
    sc[k] = s;
    mx = fmaxf(mx, s);
  }
  red[tid] = mx; __syncthreads();
  for (int off = 128; off > 0; off >>= 1) {
    if (tid < off) red[tid] = fmaxf(red[tid], red[tid + off]);
    __syncthreads();
  }
  float gm = red[0]; __syncthreads();

  float ls = 0.f;
  for (int k = tid; k < Ss; k += 256) {
    float e = __expf(sc[k] - gm);
    sc[k] = e; ls += e;
  }
  red[tid] = ls; __syncthreads();
  for (int off = 128; off > 0; off >>= 1) {
    if (tid < off) red[tid] += red[tid + off];
    __syncthreads();
  }
  float tot = red[0]; __syncthreads();

  int d = tid & 63, g = tid >> 6;   // 4 groups x 64 dims
  float pa = 0.f;
  for (int k = g; k < Ss; k += 4)
    pa += sc[k] * V[base + (size_t)k * Dd + d];
  red[tid] = pa; __syncthreads();
  if (tid < 64) {
    float o = red[tid] + red[tid + 64] + red[tid + 128] + red[tid + 192];
    CTX[base + tid] = (__bf16)(o / tot);
  }
}

// ---------------- time attention: thread per (b,h,t,p); keys = {cls, 8 temporal} ----------------
__global__ void __launch_bounds__(256) attn_time_kernel(const float* __restrict__ Q,
                                                        const float* __restrict__ K,
                                                        const float* __restrict__ V,
                                                        __bf16* __restrict__ CTX)
{
  int idx = blockIdx.x * 256 + threadIdx.x;
  const int total = Bb * Hh * Tt * Pp;
  if (idx >= total) return;
  int p = idx % Pp; int r = idx / Pp;
  int t = r % Tt;   r /= Tt;
  int h = r % Hh;   int b = r / Hh;

  const size_t hb = ((size_t)b * Ss) * Dd + (size_t)h * Dh;
  int s_q = 1 + t * Pp + p;
  const float* qp = Q + hb + (size_t)s_q * Dd;
  float q[Dh];
  #pragma unroll
  for (int d = 0; d < Dh; d++) q[d] = qp[d];

  float sc[Tt + 1];
  float mx = -1e30f;
  #pragma unroll
  for (int j = 0; j <= Tt; j++) {
    int s_k = (j == 0) ? 0 : (1 + (j - 1) * Pp + p);
    const float* kp = K + hb + (size_t)s_k * Dd;
    float s = 0.f;
    #pragma unroll
    for (int d = 0; d < Dh; d++) s += q[d] * kp[d];
    s *= 0.125f;
    sc[j] = s;
    mx = fmaxf(mx, s);
  }
  float sum = 0.f;
  #pragma unroll
  for (int j = 0; j <= Tt; j++) { sc[j] = __expf(sc[j] - mx); sum += sc[j]; }
  float inv = 1.f / sum;

  float o[Dh];
  #pragma unroll
  for (int d = 0; d < Dh; d++) o[d] = 0.f;
  #pragma unroll
  for (int j = 0; j <= Tt; j++) {
    int s_k = (j == 0) ? 0 : (1 + (j - 1) * Pp + p);
    const float* vp = V + hb + (size_t)s_k * Dd;
    float a = sc[j];
    #pragma unroll
    for (int d = 0; d < Dh; d++) o[d] += a * vp[d];
  }
  __bf16* op = CTX + hb + (size_t)s_q * Dd;
  #pragma unroll
  for (int d = 0; d < Dh; d++) op[d] = (__bf16)(o[d] * inv);
}

// ---------------- space attention: block per (b,h,t); thread per query p; flash softmax ----------------
__global__ void __launch_bounds__(256) attn_space_kernel(const float* __restrict__ Q,
                                                         const float* __restrict__ K,
                                                         const float* __restrict__ V,
                                                         __bf16* __restrict__ CTX)
{
  int t = blockIdx.x % Tt;
  int h = (blockIdx.x / Tt) % Hh;
  int b = blockIdx.x / (Tt * Hh);
  int tid = threadIdx.x;
  const size_t hb = ((size_t)b * Ss) * Dd + (size_t)h * Dh;

  __shared__ __align__(16) float sK[64 * Dh];
  __shared__ __align__(16) float sV[64 * Dh];

  const bool qv = tid < Pp;
  int s_q = 1 + t * Pp + tid;
  float q[Dh];
  if (qv) {
    const float* qp = Q + hb + (size_t)s_q * Dd;
    #pragma unroll
    for (int d = 0; d < Dh; d++) q[d] = qp[d];
  }
  float accv[Dh];
  #pragma unroll
  for (int d = 0; d < Dh; d++) accv[d] = 0.f;
  float m = -1e30f, l = 0.f;

  const int NK = Pp + 1;   // 197 keys: cls + all patches of frame t
  for (int j0 = 0; j0 < NK; j0 += 64) {
    int cnt = min(64, NK - j0);
    __syncthreads();
    for (int i = tid; i < cnt * Dh; i += 256) {
      int j = i >> 6, d = i & 63;
      int jj = j0 + j;
      int s_k = (jj == 0) ? 0 : (1 + t * Pp + jj - 1);
      sK[i] = K[hb + (size_t)s_k * Dd + d];
      sV[i] = V[hb + (size_t)s_k * Dd + d];
    }
    __syncthreads();
    if (qv) {
      for (int j = 0; j < cnt; j++) {
        float s = 0.f;
        #pragma unroll
        for (int d = 0; d < Dh; d++) s += q[d] * sK[j * Dh + d];
        s *= 0.125f;
        float mn = fmaxf(m, s);
        float c  = __expf(m - mn);
        float e  = __expf(s - mn);
        l = l * c + e;
        #pragma unroll
        for (int d = 0; d < Dh; d++) accv[d] = accv[d] * c + e * sV[j * Dh + d];
        m = mn;
      }
    }
  }
  if (qv) {
    float inv = 1.f / l;
    __bf16* op = CTX + hb + (size_t)s_q * Dd;
    #pragma unroll
    for (int d = 0; d < Dh; d++) op[d] = (__bf16)(accv[d] * inv);
  }
}

// ---------------- host orchestration ----------------
extern "C" void kernel_launch(void* const* d_in, const int* in_sizes, int n_in,
                              void* d_out, int out_size, void* d_ws, size_t ws_size,
                              hipStream_t stream) {
  (void)in_sizes; (void)n_in; (void)out_size; (void)ws_size;
  const float* x_in = (const float*)d_in[0];
  const float* tWq = (const float*)d_in[1];  const float* tbq = (const float*)d_in[2];
  const float* tWk = (const float*)d_in[3];  const float* tbk = (const float*)d_in[4];
  const float* tWv = (const float*)d_in[5];  const float* tbv = (const float*)d_in[6];
  const float* tWo = (const float*)d_in[7];  const float* tbo = (const float*)d_in[8];
  const float* sWq = (const float*)d_in[9];  const float* sbq = (const float*)d_in[10];
  const float* sWk = (const float*)d_in[11]; const float* sbk = (const float*)d_in[12];
  const float* sWv = (const float*)d_in[13]; const float* sbv = (const float*)d_in[14];
  const float* sWo = (const float*)d_in[15]; const float* sbo = (const float*)d_in[16];
  const float* W1  = (const float*)d_in[17]; const float* b1  = (const float*)d_in[18];
  const float* W2  = (const float*)d_in[19]; const float* b2  = (const float*)d_in[20];
  const float* ln1g = (const float*)d_in[21]; const float* ln1b = (const float*)d_in[22];
  const float* ln2g = (const float*)d_in[23]; const float* ln2b = (const float*)d_in[24];
  const float* ln3g = (const float*)d_in[25]; const float* ln3b = (const float*)d_in[26];

  float* X = (float*)d_out;                 // residual accumulator, also final output
  const size_t MD = (size_t)Mtot * Dd;      // 9,639,936 elems

  char* ws = (char*)d_ws;
  __bf16* Abuf = (__bf16*)ws;                       // MD bf16 (LN output)
  float*  Qb   = (float*)(ws + MD * 2);             // MD f32
  float*  Kb   = Qb + MD;                           // MD f32
  float*  Vb   = Kb + MD;                           // MD f32
  __bf16* CTXb = (__bf16*)(Vb + MD);                // MD bf16
  __bf16* Hbuf = (__bf16*)Qb;                       // Mtot x 3072 bf16 overlays dead Q+K (MD*8 bytes)
  __bf16* wts  = CTXb + MD;
  __bf16* tWqT = wts;
  __bf16* tWkT = tWqT + (size_t)Dd * Dd;
  __bf16* tWvT = tWkT + (size_t)Dd * Dd;
  __bf16* tWoT = tWvT + (size_t)Dd * Dd;
  __bf16* sWqT = tWoT + (size_t)Dd * Dd;
  __bf16* sWkT = sWqT + (size_t)Dd * Dd;
  __bf16* sWvT = sWkT + (size_t)Dd * Dd;
  __bf16* sWoT = sWvT + (size_t)Dd * Dd;
  __bf16* W1T  = sWoT + (size_t)Dd * Dd;
  __bf16* W2T  = W1T  + (size_t)Dd * Dff;

  dim3 wb(32, 8);
  dim3 wgDD(Dd / 32, Dd / 32);
  wcvt_kernel<<<wgDD, wb, 0, stream>>>(tWq, tWqT, Dd, Dd);
  wcvt_kernel<<<wgDD, wb, 0, stream>>>(tWk, tWkT, Dd, Dd);
  wcvt_kernel<<<wgDD, wb, 0, stream>>>(tWv, tWvT, Dd, Dd);
  wcvt_kernel<<<wgDD, wb, 0, stream>>>(tWo, tWoT, Dd, Dd);
  wcvt_kernel<<<wgDD, wb, 0, stream>>>(sWq, sWqT, Dd, Dd);
  wcvt_kernel<<<wgDD, wb, 0, stream>>>(sWk, sWkT, Dd, Dd);
  wcvt_kernel<<<wgDD, wb, 0, stream>>>(sWv, sWvT, Dd, Dd);
  wcvt_kernel<<<wgDD, wb, 0, stream>>>(sWo, sWoT, Dd, Dd);
  wcvt_kernel<<<dim3(Dd / 32, Dff / 32), wb, 0, stream>>>(W1, W1T, Dd, Dff);
  wcvt_kernel<<<dim3(Dff / 32, Dd / 32), wb, 0, stream>>>(W2, W2T, Dff, Dd);

  // X = hidden_states
  copy_kernel<<<(int)((MD / 4 + 255) / 256), 256, 0, stream>>>(x_in, X, (int)(MD / 4));

  dim3 g768((Mtot + 127) / 128, Dd / 128);
  dim3 gFF((Mtot + 127) / 128, Dff / 128);

  // ===== block 1: divided attention (time) =====
  ln_kernel<<<Mtot, 256, 0, stream>>>(X, ln1g, ln1b, Abuf);
  gemm_bf16_kernel<<<g768, 256, 0, stream>>>(Abuf, tWqT, tbq, nullptr, Qb, Mtot, Dd, Dd, 0);
  gemm_bf16_kernel<<<g768, 256, 0, stream>>>(Abuf, tWkT, tbk, nullptr, Kb, Mtot, Dd, Dd, 0);
  gemm_bf16_kernel<<<g768, 256, 0, stream>>>(Abuf, tWvT, tbv, nullptr, Vb, Mtot, Dd, Dd, 0);
  attn_cls_kernel<<<Bb * Hh, 256, 0, stream>>>(Qb, Kb, Vb, CTXb);
  attn_time_kernel<<<(Bb * Hh * Tt * Pp + 255) / 256, 256, 0, stream>>>(Qb, Kb, Vb, CTXb);
  gemm_bf16_kernel<<<g768, 256, 0, stream>>>(CTXb, tWoT, tbo, X, X, Mtot, Dd, Dd, 1);

  // ===== block 2: divided attention (space) =====
  ln_kernel<<<Mtot, 256, 0, stream>>>(X, ln2g, ln2b, Abuf);
  gemm_bf16_kernel<<<g768, 256, 0, stream>>>(Abuf, sWqT, sbq, nullptr, Qb, Mtot, Dd, Dd, 0);
  gemm_bf16_kernel<<<g768, 256, 0, stream>>>(Abuf, sWkT, sbk, nullptr, Kb, Mtot, Dd, Dd, 0);
  gemm_bf16_kernel<<<g768, 256, 0, stream>>>(Abuf, sWvT, sbv, nullptr, Vb, Mtot, Dd, Dd, 0);
  attn_cls_kernel<<<Bb * Hh, 256, 0, stream>>>(Qb, Kb, Vb, CTXb);
  attn_space_kernel<<<Bb * Hh * Tt, 256, 0, stream>>>(Qb, Kb, Vb, CTXb);
  gemm_bf16_kernel<<<g768, 256, 0, stream>>>(CTXb, sWoT, sbo, X, X, Mtot, Dd, Dd, 1);

  // ===== FFN =====
  ln_kernel<<<Mtot, 256, 0, stream>>>(X, ln3g, ln3b, Abuf);
  gemm_bf16_kernel<<<gFF, 256, 0, stream>>>(Abuf, W1T, b1, nullptr, Hbuf, Mtot, Dff, Dd, 2);
  gemm_bf16_kernel<<<g768, 256, 0, stream>>>(Hbuf, W2T, b2, X, X, Mtot, Dd, Dff, 1);
}
